// MultiHeadAttentionOptimised_2937757630683
// MI455X (gfx1250) — matrix operand order repaired and checked by reading, not compiled, not run on hardware
//
#include <hip/hip_runtime.h>
#include <math.h>

// ---------------------------------------------------------------------------
// MI455X (gfx1250, wave32) multi-head attention:
//   qkv = x @ W_qkv ; flash-attention per (b,h) with online softmax.
// bf16 WMMA (v_wmma_f32_16x16x32_bf16) for all matrix math.
// Double-buffered tile staging via GLOBAL_LOAD_ASYNC_TO_LDS_B128 (ASYNCcnt),
// software-pipelined so copies overlap WMMA work.
// ---------------------------------------------------------------------------

typedef __attribute__((ext_vector_type(16))) __bf16 v16bf;
typedef __attribute__((ext_vector_type(8)))  __bf16 v8bf;
typedef __attribute__((ext_vector_type(4)))  __bf16 v4bf;
typedef __attribute__((ext_vector_type(8)))  float  v8f;
typedef __attribute__((ext_vector_type(4)))  float  v4f;

#define D_MODEL   1024
#define N_HEADS   16
#define HEAD_DIM  64
#define SEQ       2048
#define BATCH     2
#define BN        (BATCH * SEQ)      // 4096 rows
#define QKV_N     (3 * D_MODEL)      // 3072
#define QSCALE    0.03125f           // 1/sqrt(D_MODEL) = 1/32

static __device__ inline v16bf pack16(v8bf lo, v8bf hi) {
    v16bf r;
#pragma unroll
    for (int i = 0; i < 8; ++i) { r[i] = lo[i]; r[i + 8] = hi[i]; }
    return r;
}

static __device__ inline v8f wmma_bf16(v16bf a, v16bf b, v8f c) {
    return __builtin_amdgcn_wmma_f32_16x16x32_bf16(
        /*neg_a=*/false, a, /*neg_b=*/false, b,
        /*c_mod=*/(short)0, c, /*reuse_a=*/false, /*reuse_b=*/false);
}

// Async global->LDS copy of 16 bytes per lane (GVS addressing: saddr + voffset).
// LDS dest address = low 32 bits of the generic pointer (ISA 10.2 aperture rule).
static __device__ inline void async_b128_to_lds(const void* sbase,
                                                unsigned voff_bytes,
                                                void* lds_ptr) {
    unsigned lds_addr = (unsigned)(size_t)lds_ptr;
    asm volatile("global_load_async_to_lds_b128 %0, %1, %2 offset:0"
                 :
                 : "v"(lds_addr), "v"(voff_bytes), "s"(sbase)
                 : "memory");
}

static __device__ inline void wait_asynccnt0() {
    asm volatile("s_wait_asynccnt 0x0" ::: "memory");
}

// ---------------------------------------------------------------------------
// Kernel 1: fp32 -> bf16 convert of x (4M elements), 4 per thread.
// ---------------------------------------------------------------------------
__global__ __launch_bounds__(256) void k_convert_x(
    const float* __restrict__ x, __bf16* __restrict__ xb, int n) {
    int i = (blockIdx.x * blockDim.x + threadIdx.x) * 4;
    if (i + 3 < n) {
        v4f v = *(const v4f*)(x + i);
        v4bf o;
#pragma unroll
        for (int e = 0; e < 4; ++e) o[e] = (__bf16)v[e];
        *(v4bf*)(xb + i) = o;
    }
}

// ---------------------------------------------------------------------------
// Kernel 2: W_qkv (1024 x 3072, row-major) -> Wt (3072 x 1024) bf16.
// ---------------------------------------------------------------------------
__global__ __launch_bounds__(256) void k_transpose_w(
    const float* __restrict__ W, __bf16* __restrict__ Wt) {
    int idx = blockIdx.x * blockDim.x + threadIdx.x;   // over 3072*1024
    if (idx >= QKV_N * D_MODEL) return;
    int k = idx & (D_MODEL - 1);
    int n = idx >> 10;
    Wt[idx] = (__bf16)W[k * QKV_N + n];
}

// ---------------------------------------------------------------------------
// Kernel 3: QKV GEMM. C[4096 x 3072] = xb[4096 x 1024] * W[1024 x 3072].
// Block tile 64(M) x 128(N), 8 waves, each wave 32x32 (2x2 WMMA tiles).
// K chunks of 32, double-buffered async-to-LDS staging pipelined vs WMMA.
// Writes Q (scaled 1/32) and K as bf16 [4096][1024];
// writes V pre-transposed per head: Vt[b*16+h][d][token] (bf16).
// ---------------------------------------------------------------------------
__global__ __launch_bounds__(256) void k_qkv_gemm(
    const __bf16* __restrict__ A,    // [BN][D_MODEL]
    const __bf16* __restrict__ Wt,   // [QKV_N][D_MODEL]
    __bf16* __restrict__ Qo, __bf16* __restrict__ Ko,
    __bf16* __restrict__ Vtg) {      // [32][HEAD_DIM][SEQ]
    __shared__ __align__(32) __bf16 As[2][64][32];    // 8 KB
    __shared__ __align__(32) __bf16 Bs[2][128][32];   // 16 KB

    const int t     = threadIdx.x;
    const int wid   = t >> 5;
    const int lane  = t & 31;
    const int lh    = lane >> 4;    // lane half (wave32)
    const int ll    = lane & 15;
    const int mBase = blockIdx.x * 64;
    const int nBase = blockIdx.y * 128;
    const int waveM = wid >> 2;     // 0..1  -> 32 rows
    const int waveN = wid & 3;      // 0..3  -> 32 cols

    // Per-thread staging coordinates (fixed for all k-chunks).
    const int ar = t >> 2, ac = (t & 3) * 8;        // A: 64 rows x 4 chunks
    const int bn = t >> 1, bc = (t & 1) * 16;       // B: 128 rows x 2 chunks

    auto stage = [&](int bufi, int k0) {
        async_b128_to_lds(A, (unsigned)(((mBase + ar) * D_MODEL + k0 + ac) * 2),
                          &As[bufi][ar][ac]);
        unsigned go = (unsigned)(((nBase + bn) * D_MODEL + k0 + bc) * 2);
        async_b128_to_lds(Wt, go,      &Bs[bufi][bn][bc]);
        async_b128_to_lds(Wt, go + 16, &Bs[bufi][bn][bc + 8]);
    };

    v8f acc[2][2];
    const v8f vzero = {};
#pragma unroll
    for (int i = 0; i < 2; ++i)
#pragma unroll
        for (int j = 0; j < 2; ++j) acc[i][j] = vzero;

    const int kb  = lh * 8;    // unused; A-frag also uses kb2
    const int kb2 = lh * 16;   // B-frag K-group base
    const int nIter = D_MODEL / 32;

    stage(0, 0);
    for (int it = 0; it < nIter; ++it) {
        const int bufi = it & 1;
        wait_asynccnt0();
        __syncthreads();
        if (it + 1 < nIter) stage(bufi ^ 1, (it + 1) * 32);  // overlap next copy

        v16bf afr[2], bfr[2];
#pragma unroll
        for (int i = 0; i < 2; ++i) {
            int row = waveM * 32 + i * 16 + ll;
            afr[i] = pack16(*(const v8bf*)&As[bufi][row][kb2],
                            *(const v8bf*)&As[bufi][row][kb2 + 8]);
        }
#pragma unroll
        for (int j = 0; j < 2; ++j) {
            int n = waveN * 32 + j * 16 + ll;
            bfr[j] = *(const v16bf*)&Bs[bufi][n][kb2];
        }
#pragma unroll
        for (int i = 0; i < 2; ++i)
#pragma unroll
            for (int j = 0; j < 2; ++j)
                acc[i][j] = wmma_bf16(afr[i], bfr[j], acc[i][j]);
    }

    // Epilogue: whole 128-wide N tile lies inside exactly one of Q/K/V.
    const int buf = nBase >> 10;         // 0=Q, 1=K, 2=V
    const int ncb = nBase & (D_MODEL - 1);
    if (buf < 2) {
        __bf16* Out    = (buf == 0) ? Qo : Ko;
        const float sc = (buf == 0) ? QSCALE : 1.0f;
#pragma unroll
        for (int i = 0; i < 2; ++i)
#pragma unroll
            for (int j = 0; j < 2; ++j) {
                int ncol = ncb + waveN * 32 + j * 16 + ll;
#pragma unroll
                for (int r = 0; r < 8; ++r) {
                    int m = mBase + waveM * 32 + i * 16 + r + lh * 8;
                    Out[m * D_MODEL + ncol] = (__bf16)(acc[i][j][r] * sc);
                }
            }
    } else {
        // V, stored transposed per head: Vtg[(b*16+h)*64 + d][token].
#pragma unroll
        for (int i = 0; i < 2; ++i)
#pragma unroll
            for (int j = 0; j < 2; ++j) {
                int ncol = ncb + waveN * 32 + j * 16 + ll;
                int h = ncol >> 6, d = ncol & (HEAD_DIM - 1);
#pragma unroll
                for (int r = 0; r < 8; ++r) {
                    int m = mBase + waveM * 32 + i * 16 + r + lh * 8;
                    int bb = m >> 11, ntok = m & (SEQ - 1);
                    Vtg[(size_t)(((bb << 4) | h) * HEAD_DIM + d) * SEQ + ntok] =
                        (__bf16)acc[i][j][r];
                }
            }
    }
}

// ---------------------------------------------------------------------------
// Kernel 4: flash attention. One block = 128 query rows of one (b,h).
// Each of 8 waves owns 16 query rows => softmax stats are wave-private.
// K and (pre-transposed) V streamed via double-buffered async-to-LDS,
// pipelined against WMMA + softmax of the previous key block.
// ---------------------------------------------------------------------------
__global__ __launch_bounds__(256) void k_flash_attn(
    const __bf16* __restrict__ Q,    // [BN][D_MODEL], head h at cols h*64..
    const __bf16* __restrict__ K,    // [BN][D_MODEL]
    const __bf16* __restrict__ Vtg,  // [32][HEAD_DIM][SEQ]
    float* __restrict__ Out) {       // [BN][D_MODEL] fp32
    __shared__ __align__(32) __bf16 Ks[2][32][HEAD_DIM];   // 8 KB (key-major)
    __shared__ __align__(32) __bf16 Vt[2][HEAD_DIM][32];   // 8 KB (d-major)
    __shared__ __align__(32) __bf16 Ps[8][16][32];         // 8 KB (per-wave P)

    const int t    = threadIdx.x;
    const int wid  = t >> 5;
    const int lane = t & 31;
    const int lh   = lane >> 4;
    const int ll   = lane & 15;
    const int bh   = blockIdx.y;            // 0..31
    const int b    = bh >> 4, h = bh & 15;
    const int qBase = blockIdx.x * 128;
    const size_t headOff = (size_t)(b * SEQ) * D_MODEL + h * HEAD_DIM;
    const int myRow = qBase + wid * 16 + ll;   // this lane's A-frag row
    const int kb  = lh * 8;                    // unused; A-frag also uses kb2
    const int kb2 = lh * 16;                   // B-frag K-group base

    // Per-thread staging coordinates.
    const int sr = t >> 3, sc = (t & 7) * 8;   // K: 32 keys x 8 chunks of 8
    const int vd = t >> 2, vc = (t & 3) * 8;   // V: 64 d x 4 chunks of 8

    auto stageKV = [&](int bufi, int j0) {
        async_b128_to_lds(K,
            (unsigned)((headOff + (size_t)(j0 + sr) * D_MODEL + sc) * 2),
            &Ks[bufi][sr][sc]);
        async_b128_to_lds(Vtg,
            (unsigned)(((bh * HEAD_DIM + vd) * SEQ + j0 + vc) * 2),
            &Vt[bufi][vd][vc]);
        // Prefetch one block further (gfx1250 global_prefetch_b8 path).
        if (j0 + 32 < SEQ)
            __builtin_prefetch(&K[headOff + (size_t)(j0 + 32 + sr) * D_MODEL], 0, 0);
    };

    // Load the wave's Q fragments once (2 d-chunks of 32).
    v16bf qf[2];
#pragma unroll
    for (int c = 0; c < 2; ++c) {
        const __bf16* qp = Q + headOff + (size_t)myRow * D_MODEL + c * 32;
        qf[c] = pack16(*(const v8bf*)(qp + kb2), *(const v8bf*)(qp + kb2 + 8));
    }

    v8f o[4];
    const v8f vzero = {};
#pragma unroll
    for (int dt = 0; dt < 4; ++dt) o[dt] = vzero;
    float m_run[8], l_run[8];
#pragma unroll
    for (int r = 0; r < 8; ++r) { m_run[r] = -1e30f; l_run[r] = 0.0f; }

    const int nIter = SEQ / 32;
    stageKV(0, 0);
    for (int it = 0; it < nIter; ++it) {
        const int bufi = it & 1;
        wait_asynccnt0();
        __syncthreads();
        if (it + 1 < nIter) stageKV(bufi ^ 1, (it + 1) * 32);  // overlap copy

        // S = Q x K^T (16 rows x 32 keys) : 2 key tiles, 2 d-chunks.
        v8f s[2];
        s[0] = vzero; s[1] = vzero;
#pragma unroll
        for (int jt = 0; jt < 2; ++jt) {
            int key = jt * 16 + ll;
#pragma unroll
            for (int c = 0; c < 2; ++c) {
                int dc = c * 32 + kb2;
                v16bf bf = *(const v16bf*)&Ks[bufi][key][dc];
                s[jt] = wmma_bf16(qf[c], bf, s[jt]);
            }
        }

        // Online softmax (rows live in 16-lane halves; reduce via shfl_xor).
        float p0[8], p1[8], csc[8];
#pragma unroll
        for (int r = 0; r < 8; ++r) {
            float v = fmaxf(s[0][r], s[1][r]);
#pragma unroll
            for (int off = 1; off < 16; off <<= 1)
                v = fmaxf(v, __shfl_xor(v, off, 32));
            float mnew = fmaxf(m_run[r], v);
            csc[r] = __expf(m_run[r] - mnew);
            p0[r]  = __expf(s[0][r] - mnew);
            p1[r]  = __expf(s[1][r] - mnew);
            float rs = p0[r] + p1[r];
#pragma unroll
            for (int off = 1; off < 16; off <<= 1)
                rs += __shfl_xor(rs, off, 32);
            l_run[r] = l_run[r] * csc[r] + rs;
            m_run[r] = mnew;
        }
        // Rescale running O.
#pragma unroll
        for (int dt = 0; dt < 4; ++dt)
#pragma unroll
            for (int r = 0; r < 8; ++r) o[dt][r] *= csc[r];

        // P -> wave-private LDS (accumulator layout), then back as A-frag.
#pragma unroll
        for (int r = 0; r < 8; ++r) {
            int prow = r + lh * 8;
            Ps[wid][prow][ll]      = (__bf16)p0[r];
            Ps[wid][prow][16 + ll] = (__bf16)p1[r];
        }
        v16bf pf = pack16(*(const v8bf*)&Ps[wid][ll][kb2],
                          *(const v8bf*)&Ps[wid][ll][kb2 + 8]);

        // O += P x V  (4 output-d tiles of 16; B-frag contiguous 32 B).
#pragma unroll
        for (int dt = 0; dt < 4; ++dt) {
            int d = dt * 16 + ll;
            v16bf vf = *(const v16bf*)&Vt[bufi][d][kb2];
            o[dt] = wmma_bf16(pf, vf, o[dt]);
        }
    }

    // Epilogue: normalize and write fp32 output.
#pragma unroll
    for (int dt = 0; dt < 4; ++dt) {
        int d = dt * 16 + ll;
#pragma unroll
        for (int r = 0; r < 8; ++r) {
            int m = qBase + wid * 16 + r + lh * 8;
            Out[headOff + (size_t)m * D_MODEL + d] = o[dt][r] / l_run[r];
        }
    }
}

// ---------------------------------------------------------------------------
// Host launcher
// ---------------------------------------------------------------------------
extern "C" void kernel_launch(void* const* d_in, const int* in_sizes, int n_in,
                              void* d_out, int out_size, void* d_ws, size_t ws_size,
                              hipStream_t stream) {
    const float* x = (const float*)d_in[0];   // (2, 2048, 1024) fp32
    const float* W = (const float*)d_in[1];   // (1024, 3072)    fp32
    float* out = (float*)d_out;               // (2, 2048, 1024) fp32

    // Workspace layout (bf16), ~38 MB total:
    char* ws = (char*)d_ws;
    size_t off = 0;
    __bf16* xb  = (__bf16*)(ws + off); off += (size_t)BN * D_MODEL * 2;     // 8 MB
    __bf16* Wt  = (__bf16*)(ws + off); off += (size_t)QKV_N * D_MODEL * 2;  // 6 MB
    __bf16* Qb  = (__bf16*)(ws + off); off += (size_t)BN * D_MODEL * 2;     // 8 MB
    __bf16* Kb  = (__bf16*)(ws + off); off += (size_t)BN * D_MODEL * 2;     // 8 MB
    __bf16* Vtg = (__bf16*)(ws + off); off += (size_t)BN * D_MODEL * 2;     // 8 MB
    (void)ws_size; (void)in_sizes; (void)n_in; (void)out_size;

    // 1) x -> bf16
    {
        int n = BN * D_MODEL;
        int threads = 256, blocks = (n / 4 + threads - 1) / threads;
        k_convert_x<<<blocks, threads, 0, stream>>>(x, xb, n);
    }
    // 2) W -> Wt (transposed bf16)
    {
        int n = QKV_N * D_MODEL;
        int threads = 256, blocks = (n + threads - 1) / threads;
        k_transpose_w<<<blocks, threads, 0, stream>>>(W, Wt);
    }
    // 3) QKV GEMM (Q, K row-major; V transposed per head)
    {
        dim3 grid(BN / 64, QKV_N / 128);   // (64, 24)
        k_qkv_gemm<<<grid, 256, 0, stream>>>(xb, Wt, Qb, Kb, Vtg);
    }
    // 4) Flash attention
    {
        dim3 grid(SEQ / 128, BATCH * N_HEADS);   // (16, 32)
        k_flash_attn<<<grid, 256, 0, stream>>>(Qb, Kb, Vtg, out);
    }
}
